// GAT_2894807958002
// MI455X (gfx1250) — compile-verified, matrix operand order checked by reading, not run on hardware
//
#include <hip/hip_runtime.h>

// ---------------- problem constants (match reference) ----------------
#define NNODES 50000
#define NEDGES 640000
#define HID    128
#define HEADS  2
#define CPH    64          // channels per head
#define ATT_SLOPE 0.2f
#define OUT_SLOPE 0.01f

typedef __attribute__((ext_vector_type(16))) _Float16 v16h;
typedef __attribute__((ext_vector_type(8)))  _Float16 v8h;
typedef __attribute__((ext_vector_type(8)))  float    v8f;

__device__ __forceinline__ float lrelu(float v, float s) { return v > 0.f ? v : v * s; }

// order-preserving float <-> int key for integer atomicMax
__device__ __forceinline__ int   fkey(float v) { int b = __float_as_int(v); return b >= 0 ? b : (b ^ 0x7fffffff); }
__device__ __forceinline__ float fdec(int k)   { return __int_as_float(k >= 0 ? k : (k ^ 0x7fffffff)); }

// ---------------------------------------------------------------------
// f32 -> f16 pack: one thread converts 8 floats (two float4 loads, one b128 store)
// ---------------------------------------------------------------------
__global__ __launch_bounds__(256) void cvt_f32_f16(const float* __restrict__ src,
                                                   _Float16* __restrict__ dst, int n8)
{
    int i = blockIdx.x * blockDim.x + threadIdx.x;
    if (i >= n8) return;
    const float4* s = (const float4*)src;
    float4 a = s[2 * i], b = s[2 * i + 1];
    v8h o;
    o[0] = (_Float16)a.x; o[1] = (_Float16)a.y; o[2] = (_Float16)a.z; o[3] = (_Float16)a.w;
    o[4] = (_Float16)b.x; o[5] = (_Float16)b.y; o[6] = (_Float16)b.z; o[7] = (_Float16)b.w;
    ((v8h*)dst)[i] = o;
}

// ---------------------------------------------------------------------
// Y[N,128] = Xh[N,128] @ Wh^T  (f16 in, f32 out), W torch layout [out=128,in=128].
// One wave per 16-row slab: preload 4 A fragments (K=128), then loop the 8
// column tiles; per tile: 4 x (2 b128 loads of B) + 4 x v_wmma, store 16x16.
// A/B lane layout per ISA 7.12.2: lanes 0-15 row M=lane, halves = K
// {k0..k0+7, k0+16..k0+23}; lanes 16-31 row M=lane-16 with k0 += 8.
// ---------------------------------------------------------------------
__global__ __launch_bounds__(32) void gemm_xwT_wmma(const _Float16* __restrict__ Xh,
                                                    const _Float16* __restrict__ Wh,
                                                    float* __restrict__ Y)
{
    const int mtile = blockIdx.x;          // 3125 row tiles (50000/16)
    const int lane  = threadIdx.x;
    const int half  = lane >> 4;
    const int l15   = lane & 15;

    // ---- preload A fragments (X read exactly once per GEMM) ----
    const v8h* xq = (const v8h*)(Xh + (size_t)(mtile * 16 + l15) * HID);
    v16h A[4];
#pragma unroll
    for (int kk = 0; kk < 4; ++kk) {
        v8h lo = xq[kk * 4 + half];        // K = k0 .. k0+7
        v8h hi = xq[kk * 4 + half + 2];    // K = k0+16 .. k0+23
#pragma unroll
        for (int i = 0; i < 8; ++i) { A[kk][i] = lo[i]; A[kk][i + 8] = hi[i]; }
    }

    // ---- 8 column tiles, one live accumulator ----
#pragma unroll
    for (int n = 0; n < 8; ++n) {
        const v8h* wq = (const v8h*)(Wh + (size_t)(n * 16 + l15) * HID);
        v8f acc = {0.f, 0.f, 0.f, 0.f, 0.f, 0.f, 0.f, 0.f};
#pragma unroll
        for (int kk = 0; kk < 4; ++kk) {
            v8h lo = wq[kk * 4 + half];
            v8h hi = wq[kk * 4 + half + 2];
            v16h B;
#pragma unroll
            for (int i = 0; i < 8; ++i) { B[i] = lo[i]; B[i + 8] = hi[i]; }
            acc = __builtin_amdgcn_wmma_f32_16x16x32_f16(false, A[kk], false, B,
                                                         (short)0, acc, false, false);
        }
        // D layout: VGPR r -> row (r + 8*half), col l15
        float* y = Y + (size_t)(mtile * 16 + half * 8) * HID + n * 16 + l15;
#pragma unroll
        for (int r = 0; r < 8; ++r) y[(size_t)r * HID] = acc[r];
    }
}

// ---------------------------------------------------------------------
__global__ __launch_bounds__(256) void init_layer(int* __restrict__ amax,
                                                  float* __restrict__ denom,
                                                  float* __restrict__ agg,
                                                  int nnd, int nagg)
{
    int i = blockIdx.x * blockDim.x + threadIdx.x;
    if (i < nnd) { amax[i] = (int)0x80000000; denom[i] = 0.f; }
    if (i < nagg) agg[i] = 0.f;
}

// pass 1: per (edge, head): alpha = att . leaky_relu(xl[src]+xr[dst]); segment max
__global__ __launch_bounds__(256) void edge_alpha(const float* __restrict__ xl,
                                                  const float* __restrict__ xr,
                                                  const int* __restrict__ ei,
                                                  const float* __restrict__ att,
                                                  float* __restrict__ alpha,
                                                  int* __restrict__ amax)
{
    int t = blockIdx.x * blockDim.x + threadIdx.x;
    if (t >= NEDGES * HEADS) return;
    int e = t >> 1, h = t & 1;
    int s = ei[e], d = ei[NEDGES + e];
    const float4* xs = (const float4*)(xl + (size_t)s * HID + h * CPH);
    const float4* xd = (const float4*)(xr + (size_t)d * HID + h * CPH);
    const float4* at = (const float4*)(att + h * CPH);
    float acc = 0.f;
#pragma unroll
    for (int i = 0; i < CPH / 4; ++i) {
        float4 a = xs[i], b = xd[i], w = at[i];
        acc += w.x * lrelu(a.x + b.x, ATT_SLOPE);
        acc += w.y * lrelu(a.y + b.y, ATT_SLOPE);
        acc += w.z * lrelu(a.z + b.z, ATT_SLOPE);
        acc += w.w * lrelu(a.w + b.w, ATT_SLOPE);
    }
    alpha[t] = acc;
    atomicMax(amax + d * HEADS + h, fkey(acc));
}

// pass 2: ex = exp(alpha - amax[dst]); segment sum of ex
__global__ __launch_bounds__(256) void edge_expsum(const int* __restrict__ ei,
                                                   const int* __restrict__ amax,
                                                   float* __restrict__ alpha,
                                                   float* __restrict__ denom)
{
    int t = blockIdx.x * blockDim.x + threadIdx.x;
    if (t >= NEDGES * HEADS) return;
    int e = t >> 1, h = t & 1;
    int d = ei[NEDGES + e];
    float m  = fdec(amax[d * HEADS + h]);
    float ex = __expf(alpha[t] - m);
    alpha[t] = ex;
    atomicAdd(denom + d * HEADS + h, ex);
}

// pass 3: agg[dst] += (ex/denom[dst]) * xl[src]; 1 thread = 1 float4 chunk of one edge
__global__ __launch_bounds__(256) void edge_agg(const float* __restrict__ xl,
                                                const int* __restrict__ ei,
                                                const float* __restrict__ alpha,
                                                const float* __restrict__ denom,
                                                float* __restrict__ agg)
{
    int t = blockIdx.x * blockDim.x + threadIdx.x;
    if (t >= NEDGES * (HID / 4)) return;
    int e = t >> 5, q = t & 31;            // 32 float4 chunks per edge
    int h = q >> 4;                        // chunks 0-15 -> head 0, 16-31 -> head 1
    int s = ei[e], d = ei[NEDGES + e];
    float w = alpha[e * HEADS + h] / (denom[d * HEADS + h] + 1e-16f);
    float4 xv = ((const float4*)xl)[(size_t)s * (HID / 4) + q];
    float* o = agg + (size_t)d * HID + q * 4;
    atomicAdd(o + 0, w * xv.x);
    atomicAdd(o + 1, w * xv.y);
    atomicAdd(o + 2, w * xv.z);
    atomicAdd(o + 3, w * xv.w);
}

__global__ __launch_bounds__(256) void finalize_h1(const float* __restrict__ agg,
                                                   const float* __restrict__ b,
                                                   float* __restrict__ h1, int n)
{
    int i = blockIdx.x * blockDim.x + threadIdx.x;
    if (i < n) h1[i] = lrelu(agg[i] + b[i & (HID - 1)], OUT_SLOPE);
}

__global__ __launch_bounds__(256) void finalize_out(const float* __restrict__ agg,
                                                    const float* __restrict__ b,
                                                    const float* __restrict__ h1,
                                                    float* __restrict__ out, int n)
{
    int i = blockIdx.x * blockDim.x + threadIdx.x;
    if (i < n) out[i] = h1[i] + (agg[i] + b[i & (HID - 1)]);
}

// ---------------------------------------------------------------------
extern "C" void kernel_launch(void* const* d_in, const int* in_sizes, int n_in,
                              void* d_out, int out_size, void* d_ws, size_t ws_size,
                              hipStream_t stream)
{
    const float* x    = (const float*)d_in[0];
    const int*   ei   = (const int*)  d_in[1];
    const float* wl1  = (const float*)d_in[2];
    const float* wr1  = (const float*)d_in[3];
    const float* att1 = (const float*)d_in[4];
    const float* b1   = (const float*)d_in[5];
    const float* wl2  = (const float*)d_in[6];
    const float* wr2  = (const float*)d_in[7];
    const float* att2 = (const float*)d_in[8];
    const float* b2   = (const float*)d_in[9];
    float* out = (float*)d_out;

    const size_t NH = (size_t)NNODES * HID;   // 6.4M
    const size_t WSZ = (size_t)HID * HID;     // 16384
    float* xl    = (float*)d_ws;              // f32 buffers, reused both layers
    float* xr    = xl    + NH;
    float* h1    = xr    + NH;                // layer-1 activation (kept for residual)
    float* agg   = h1    + NH;                // aggregation accumulator
    float* alpha = agg   + NH;                // per-(edge,head) scratch
    float* denom = alpha + (size_t)NEDGES * HEADS;
    int*   amax  = (int*)(denom + (size_t)NNODES * HEADS);
    _Float16* xh  = (_Float16*)(amax + (size_t)NNODES * HEADS);  // f16 activations
    _Float16* whl = xh  + NH;                                    // f16 weights
    _Float16* whr = whl + WSZ;
    // total ~121 MB of d_ws

    const int NND  = NNODES * HEADS;
    const int NAGG = NNODES * HID;
    const int EH   = NEDGES * HEADS;
    const int EQ   = NEDGES * (HID / 4);
    const int X8   = (int)(NH / 8);           // 8-float chunks in an activation
    const int W8   = (int)(WSZ / 8);

    dim3 gb(32);
    dim3 gg(NNODES / 16);                     // 3125 waves per GEMM
    const int TB = 256;

    // -------- layer 1 --------
    cvt_f32_f16 <<<(X8 + TB - 1) / TB, TB, 0, stream>>>(x,   xh,  X8);
    cvt_f32_f16 <<<(W8 + TB - 1) / TB, TB, 0, stream>>>(wl1, whl, W8);
    cvt_f32_f16 <<<(W8 + TB - 1) / TB, TB, 0, stream>>>(wr1, whr, W8);
    gemm_xwT_wmma<<<gg, gb, 0, stream>>>(xh, whl, xl);
    gemm_xwT_wmma<<<gg, gb, 0, stream>>>(xh, whr, xr);
    init_layer  <<<(NAGG + TB - 1) / TB, TB, 0, stream>>>(amax, denom, agg, NND, NAGG);
    edge_alpha  <<<(EH   + TB - 1) / TB, TB, 0, stream>>>(xl, xr, ei, att1, alpha, amax);
    edge_expsum <<<(EH   + TB - 1) / TB, TB, 0, stream>>>(ei, amax, alpha, denom);
    edge_agg    <<<(EQ   + TB - 1) / TB, TB, 0, stream>>>(xl, ei, alpha, denom, agg);
    finalize_h1 <<<(NAGG + TB - 1) / TB, TB, 0, stream>>>(agg, b1, h1, NAGG);

    // -------- layer 2 --------
    cvt_f32_f16 <<<(X8 + TB - 1) / TB, TB, 0, stream>>>(h1,  xh,  X8);
    cvt_f32_f16 <<<(W8 + TB - 1) / TB, TB, 0, stream>>>(wl2, whl, W8);
    cvt_f32_f16 <<<(W8 + TB - 1) / TB, TB, 0, stream>>>(wr2, whr, W8);
    gemm_xwT_wmma<<<gg, gb, 0, stream>>>(xh, whl, xl);
    gemm_xwT_wmma<<<gg, gb, 0, stream>>>(xh, whr, xr);
    init_layer  <<<(NAGG + TB - 1) / TB, TB, 0, stream>>>(amax, denom, agg, NND, NAGG);
    edge_alpha  <<<(EH   + TB - 1) / TB, TB, 0, stream>>>(xl, xr, ei, att2, alpha, amax);
    edge_expsum <<<(EH   + TB - 1) / TB, TB, 0, stream>>>(ei, amax, alpha, denom);
    edge_agg    <<<(EQ   + TB - 1) / TB, TB, 0, stream>>>(xl, ei, alpha, denom, agg);
    finalize_out<<<(NAGG + TB - 1) / TB, TB, 0, stream>>>(agg, b2, h1, out, NAGG);
}
